// MultiHeadAttention_43258910605506
// MI455X (gfx1250) — matrix operand order repaired and checked by reading, not compiled, not run on hardware
//
#include <hip/hip_runtime.h>
#include <hip/hip_bf16.h>
#include <stdint.h>

// ---------------------------------------------------------------------------
// MHA forward for B=2, S=2048, D=1024, H=16, DK=64 on gfx1250 (wave32, WMMA)
// ---------------------------------------------------------------------------

#define B_  2
#define S_  2048
#define D_  1024
#define H_  16
#define DK_ 64
#define M_  (B_ * S_)          // 4096 rows in the flattened [B*S, D] GEMMs

typedef __attribute__((ext_vector_type(16))) _Float16 v16h;
typedef __attribute__((ext_vector_type(8)))  _Float16 v8h;
typedef __attribute__((ext_vector_type(8)))  float    v8f;

#define WMMA_F16(A, Bf, C) \
  __builtin_amdgcn_wmma_f32_16x16x32_f16(false, (A), false, (Bf), (short)0, (C), false, false)

__device__ __forceinline__ v8f zero8() {
  v8f z;
#pragma unroll
  for (int j = 0; j < 8; ++j) z[j] = 0.0f;
  return z;
}

__device__ __forceinline__ v16h frag_cat(v8h lo, v8h hi) {
  v16h r;
#pragma unroll
  for (int j = 0; j < 8; ++j) { r[j] = lo[j]; r[j + 8] = hi[j]; }
  return r;
}

// A fragment (16xK=32, f16): lanes 0-15 -> M=lane, K=k0..k0+15
//                            lanes 16-31 -> M=lane-16, K=k0+16..k0+31
__device__ __forceinline__ v16h load_a_f16(const _Float16* __restrict__ base, int ld,
                                           int k0, int lane) {
  const _Float16* p = base + (size_t)(lane & 15) * ld + k0 + ((lane >> 4) << 4);
  return frag_cat(*(const v8h*)p, *(const v8h*)(p + 8));
}

// B fragment (K=32 x 16, f16): lane (&15) = N column; lanes 0-15 hold K=k0..k0+15,
// lanes 16-31 hold K=k0+16..k0+31 (16 contiguous K values per lane).
__device__ __forceinline__ v16h load_b_f16(const _Float16* __restrict__ base, int ld,
                                           int ncol0, int k0, int lane) {
  const _Float16* p = base + (size_t)(ncol0 + (lane & 15)) * ld + k0 + ((lane >> 4) << 4);
  return frag_cat(*(const v8h*)p, *(const v8h*)(p + 8));
}

// B fragment out of an LDS tile [64 rows x 32 K] f16, row-major
__device__ __forceinline__ v16h lds_b_frag(const _Float16* tile, int ncol0, int lane) {
  const _Float16* p = tile + (size_t)(ncol0 + (lane & 15)) * 32 + ((lane >> 4) << 4);
  return frag_cat(*(const v8h*)p, *(const v8h*)(p + 8));
}

// ---------------------------------------------------------------------------
// Kernel 0: fp32 -> f16 elementwise convert (n multiple of 8)
// ---------------------------------------------------------------------------
__global__ __launch_bounds__(256) void cvt_f32_f16_kernel(const float* __restrict__ src,
                                                          _Float16* __restrict__ dst,
                                                          int n) {
  const int i = (blockIdx.x * 256 + threadIdx.x) * 8;
  if (i < n) {
    v8h o;
#pragma unroll
    for (int j = 0; j < 8; ++j) o[j] = (_Float16)src[i + j];
    *(v8h*)(dst + i) = o;
  }
}

// ---------------------------------------------------------------------------
// Shared GEMM machinery: 256 threads, 8 waves; M-tile 256 (32 rows/wave),
// N-tile 64. Weight tile (64 x 32 f16, 4 KB) is DMA'd to LDS with
// GLOBAL_LOAD_ASYNC_TO_LDS_B128 (ASYNCcnt), double-buffered.
// ---------------------------------------------------------------------------
__device__ __forceinline__ void async_w_tile(const _Float16* W, int Nbase, int kk,
                                             _Float16* ldsbuf, int tid) {
  const int wrow = tid >> 2;            // 0..63  (row within N-tile)
  const int wseg = tid & 3;             // 0..3   (16B segment within 32-half row)
  const uint32_t voff = (uint32_t)(wrow * (D_ * 2) + wseg * 16);     // bytes
  const uint64_t sbase = (uint64_t)(uintptr_t)(W + (size_t)Nbase * D_ + kk);
  const uint32_t ldso = (uint32_t)(uintptr_t)ldsbuf + (uint32_t)(wrow * 64 + wseg * 16);
  asm volatile("global_load_async_to_lds_b128 %0, %1, %2"
               :: "v"(ldso), "v"(voff), "s"(sbase) : "memory");
}

__device__ __forceinline__ void wait_async_and_barrier() {
  asm volatile("s_wait_asynccnt 0x0" ::: "memory");
  __syncthreads();
}

// ---------------------------------------------------------------------------
// Kernel 1: fused Q/K/V projection (+bias) + RoPE, f16 outputs (all-f16 GEMM).
//   z=0: qh[B,H,S,DK] (rope, pre-scaled by 1/sqrt(DK))
//   z=1: kh[B,H,S,DK] (rope)      z=2: vt[B,H,DK,S]
// grid (D/64, M/256, 3), block 256
// ---------------------------------------------------------------------------
__global__ __launch_bounds__(256) void qkv_proj_rope_kernel(
    const _Float16* __restrict__ qf, const _Float16* __restrict__ kf,
    const _Float16* __restrict__ vf,
    const _Float16* __restrict__ Wqh, const _Float16* __restrict__ Wkh,
    const _Float16* __restrict__ Wvh,
    const float* __restrict__ bq, const float* __restrict__ bk,
    const float* __restrict__ bv,
    _Float16* __restrict__ qh, _Float16* __restrict__ kh, _Float16* __restrict__ vt) {
  __shared__ __align__(16) _Float16 ldsW[2][64 * 32];

  const int z = blockIdx.z;
  const _Float16* X    = (z == 0) ? qf  : (z == 1) ? kf  : vf;
  const _Float16* W    = (z == 0) ? Wqh : (z == 1) ? Wkh : Wvh;
  const float*    bias = (z == 0) ? bq  : (z == 1) ? bk  : bv;

  const int tid   = threadIdx.x;
  const int lane  = tid & 31;
  const int wv    = tid >> 5;
  const int Nbase = blockIdx.x * 64;                 // head h = blockIdx.x
  const int Mbase = blockIdx.y * 256 + wv * 32;      // 32 rows per wave

  v8f acc[2][4];
#pragma unroll
  for (int g = 0; g < 2; ++g)
#pragma unroll
    for (int t = 0; t < 4; ++t) acc[g][t] = zero8();

  const _Float16* Xrow0 = X + (size_t)Mbase * D_;
  const _Float16* Xrow1 = X + (size_t)(Mbase + 16) * D_;

  async_w_tile(W, Nbase, 0, &ldsW[0][0], tid);
  wait_async_and_barrier();

  int ibuf = 0;
  for (int kk = 0; kk < D_; kk += 32) {
    if (kk + 32 < D_) async_w_tile(W, Nbase, kk + 32, &ldsW[ibuf ^ 1][0], tid);
    v16h a0 = load_a_f16(Xrow0, D_, kk, lane);
    v16h a1 = load_a_f16(Xrow1, D_, kk, lane);
    // hoist all 4 B-fragments: one ds clause, then 8 back-to-back WMMAs
    v16h b0 = lds_b_frag(&ldsW[ibuf][0], 0,  lane);
    v16h b1 = lds_b_frag(&ldsW[ibuf][0], 16, lane);
    v16h b2 = lds_b_frag(&ldsW[ibuf][0], 32, lane);
    v16h b3 = lds_b_frag(&ldsW[ibuf][0], 48, lane);
    acc[0][0] = WMMA_F16(a0, b0, acc[0][0]);
    acc[1][0] = WMMA_F16(a1, b0, acc[1][0]);
    acc[0][1] = WMMA_F16(a0, b1, acc[0][1]);
    acc[1][1] = WMMA_F16(a1, b1, acc[1][1]);
    acc[0][2] = WMMA_F16(a0, b2, acc[0][2]);
    acc[1][2] = WMMA_F16(a1, b2, acc[1][2]);
    acc[0][3] = WMMA_F16(a0, b3, acc[0][3]);
    acc[1][3] = WMMA_F16(a1, b3, acc[1][3]);
    wait_async_and_barrier();
    ibuf ^= 1;
  }

  const int h      = blockIdx.x;
  const int rowoff = (lane >> 4) << 3;
  const int nl     = lane & 15;
#pragma unroll
  for (int g = 0; g < 2; ++g) {
#pragma unroll
    for (int t = 0; t < 4; ++t) {
      const int dk = t * 16 + nl;
      const int n  = Nbase + dk;
      const float bval = bias[n];
      const float freq = __powf(10000.0f, -(float)(dk >> 1) * (1.0f / 16.0f));
#pragma unroll
      for (int r = 0; r < 8; ++r) {
        const int row = Mbase + g * 16 + rowoff + r;
        const int s   = row & (S_ - 1);
        const int bb  = row >> 11;                   // row / S_
        float val = acc[g][t][r] + bval;
        if (z < 2) {                                 // RoPE on q and k
          float part = __shfl_xor(val, 1, 32);       // pair partner column dk^1
          float sn, cs;
          __sincosf((float)s * freq, &sn, &cs);
          val = val * cs + part * ((dk & 1) ? sn : -sn);
          if (z == 0) val *= 0.125f;                 // fold 1/sqrt(DK) into Q
        }
        const size_t bh = (size_t)bb * H_ + h;
        if (z == 2)
          vt[(bh * DK_ + dk) * S_ + s] = (_Float16)val;        // [B,H,DK,S]
        else {
          _Float16* Dp = (z == 0) ? qh : kh;
          Dp[(bh * S_ + s) * DK_ + dk] = (_Float16)val;        // [B,H,S,DK]
        }
      }
    }
  }
}

// ---------------------------------------------------------------------------
// Kernel 2: causal flash attention (transposed: S^T = K x Q^T, O^T = V^T x P^T)
// One wave per 16-query tile; query = lane, so softmax stats are per-lane VALU
// and P^T feeds WMMA as a B-fragment after one shfl_xor(16) half-merge.
// grid (B*H*S/16/4), block 128 (4 waves), no LDS.
// ---------------------------------------------------------------------------
__global__ __launch_bounds__(128) void flash_attn_kernel(
    const _Float16* __restrict__ qh, const _Float16* __restrict__ kh,
    const _Float16* __restrict__ vt, _Float16* __restrict__ cx) {
  const int lane = threadIdx.x & 31;
  const int wv   = threadIdx.x >> 5;
  const int tile = blockIdx.x * 4 + wv;                 // 0 .. B*H*S/16-1
  const int bh   = tile >> 7;                           // / (S/16)
  const int qb   = (tile & 127) << 4;

  const _Float16* Qb = qh + ((size_t)bh * S_ + qb) * DK_;
  const _Float16* Kb = kh + (size_t)bh * S_ * DK_;
  const _Float16* Vb = vt + (size_t)bh * DK_ * S_;

  const int nl     = lane & 15;
  const int rowloc = (lane >> 4) << 3;                  // 0 or 8
  const int query  = qb + nl;                           // this lane's query row
  const bool lowhalf = lane < 16;

  // Q^T as B-matrix (dk x 16 queries), loaded once
  const v16h bq0 = load_b_f16(Qb, DK_, 0, 0,  lane);
  const v16h bq1 = load_b_f16(Qb, DK_, 0, 32, lane);

  float mQ = -1e38f, lQ = 0.0f;                         // per-lane (per-query)
  v8f o[4];                                             // O^T: lane=query, vgpr=dk
#pragma unroll
  for (int t = 0; t < 4; ++t) o[t] = zero8();

  for (int kv = 0; kv < qb + 16; kv += 32) {
    // ---- S^T tiles: st0 = keys [kv..kv+15], st1 = keys [kv+16..kv+31] ----
    v8f s0 = zero8(), s1 = zero8();
    {
      v16h ak00 = load_a_f16(Kb + (size_t)kv * DK_,        DK_, 0,  lane);
      v16h ak01 = load_a_f16(Kb + (size_t)kv * DK_,        DK_, 32, lane);
      v16h ak10 = load_a_f16(Kb + (size_t)(kv + 16) * DK_, DK_, 0,  lane);
      v16h ak11 = load_a_f16(Kb + (size_t)(kv + 16) * DK_, DK_, 32, lane);
      s0 = WMMA_F16(ak00, bq0, s0);
      s0 = WMMA_F16(ak01, bq1, s0);
      s1 = WMMA_F16(ak10, bq0, s1);
      s1 = WMMA_F16(ak11, bq1, s1);
    }
    // ---- causal mask + per-lane online softmax (no cross-lane reductions) ----
    float mx = -1e38f;
#pragma unroll
    for (int r = 0; r < 8; ++r) {
      const int key0 = kv + rowloc + r;                 // key index of s0[r]
      float x0 = s0[r]; if (key0 > query)      x0 = -1e9f;
      float x1 = s1[r]; if (key0 + 16 > query) x1 = -1e9f;
      s0[r] = x0; s1[r] = x1;
      mx = fmaxf(mx, fmaxf(x0, x1));
    }
    const float mnew  = fmaxf(mQ, mx);
    const float alpha = __expf(mQ - mnew);
    mQ = mnew;
    float p0[8], p1[8], sum = 0.0f;
#pragma unroll
    for (int r = 0; r < 8; ++r) {
      p0[r] = __expf(s0[r] - mnew);
      p1[r] = __expf(s1[r] - mnew);
      sum += p0[r] + p1[r];
    }
    lQ = lQ * alpha + sum;
#pragma unroll
    for (int t = 0; t < 4; ++t)
#pragma unroll
      for (int r = 0; r < 8; ++r) o[t][r] *= alpha;
    // ---- P^T as B-fragment: merge lane halves with shfl_xor(16) ----
    v16h pb;
#pragma unroll
    for (int j = 0; j < 8; ++j) {
      const float q0p = __shfl_xor(p0[j], 16, 32);
      const float q1p = __shfl_xor(p1[j], 16, 32);
      pb[j]     = (_Float16)(lowhalf ? p0[j] : q1p);    // keys kv+[0..7] / [16..23]
      pb[j + 8] = (_Float16)(lowhalf ? q0p  : p1[j]);   // keys kv+[8..15] / [24..31]
    }
    // ---- O^T += V^T(16dk x 32keys) x P^T(32keys x 16q) ----
#pragma unroll
    for (int t = 0; t < 4; ++t) {
      v16h av = load_a_f16(Vb + (size_t)(t * 16) * S_ + kv, S_, 0, lane);
      o[t] = WMMA_F16(av, pb, o[t]);
    }
  }

  // ---- normalize + store ctx[B,S,D]: dk contiguous per lane -> b128 stores ----
  const int bb = bh >> 4, h = bh & 15;
  const float inv = 1.0f / lQ;
  const size_t base = ((size_t)bb * S_ + query) * D_ + h * DK_ + rowloc;
#pragma unroll
  for (int t = 0; t < 4; ++t) {
    v8h pack;
#pragma unroll
    for (int r = 0; r < 8; ++r) pack[r] = (_Float16)(o[t][r] * inv);
    *(v8h*)(cx + base + t * 16) = pack;
  }
}

// ---------------------------------------------------------------------------
// Kernel 3: output projection  out = ctx @ Wo^T + bo   (fp32 out)
// grid (D/64, M/256), block 256; Wo tile async-staged in LDS like kernel 1
// ---------------------------------------------------------------------------
__global__ __launch_bounds__(256) void out_proj_kernel(
    const _Float16* __restrict__ cx, const _Float16* __restrict__ Woh,
    const float* __restrict__ bo, float* __restrict__ out) {
  __shared__ __align__(16) _Float16 ldsW[2][64 * 32];

  const int tid   = threadIdx.x;
  const int lane  = tid & 31;
  const int wv    = tid >> 5;
  const int Nbase = blockIdx.x * 64;
  const int Mbase = blockIdx.y * 256 + wv * 32;

  v8f acc[2][4];
#pragma unroll
  for (int g = 0; g < 2; ++g)
#pragma unroll
    for (int t = 0; t < 4; ++t) acc[g][t] = zero8();

  const _Float16* Crow0 = cx + (size_t)Mbase * D_;
  const _Float16* Crow1 = cx + (size_t)(Mbase + 16) * D_;

  async_w_tile(Woh, Nbase, 0, &ldsW[0][0], tid);
  wait_async_and_barrier();

  int ibuf = 0;
  for (int kk = 0; kk < D_; kk += 32) {
    if (kk + 32 < D_) async_w_tile(Woh, Nbase, kk + 32, &ldsW[ibuf ^ 1][0], tid);
    v16h a0 = load_a_f16(Crow0, D_, kk, lane);
    v16h a1 = load_a_f16(Crow1, D_, kk, lane);
    v16h b0 = lds_b_frag(&ldsW[ibuf][0], 0,  lane);
    v16h b1 = lds_b_frag(&ldsW[ibuf][0], 16, lane);
    v16h b2 = lds_b_frag(&ldsW[ibuf][0], 32, lane);
    v16h b3 = lds_b_frag(&ldsW[ibuf][0], 48, lane);
    acc[0][0] = WMMA_F16(a0, b0, acc[0][0]);
    acc[1][0] = WMMA_F16(a1, b0, acc[1][0]);
    acc[0][1] = WMMA_F16(a0, b1, acc[0][1]);
    acc[1][1] = WMMA_F16(a1, b1, acc[1][1]);
    acc[0][2] = WMMA_F16(a0, b2, acc[0][2]);
    acc[1][2] = WMMA_F16(a1, b2, acc[1][2]);
    acc[0][3] = WMMA_F16(a0, b3, acc[0][3]);
    acc[1][3] = WMMA_F16(a1, b3, acc[1][3]);
    wait_async_and_barrier();
    ibuf ^= 1;
  }

  const int rowoff = (lane >> 4) << 3;
  const int nl     = lane & 15;
#pragma unroll
  for (int g = 0; g < 2; ++g)
#pragma unroll
    for (int t = 0; t < 4; ++t) {
      const int n = Nbase + t * 16 + nl;
      const float bval = bo[n];
#pragma unroll
      for (int r = 0; r < 8; ++r) {
        const int row = Mbase + g * 16 + rowoff + r;
        out[(size_t)row * D_ + n] = acc[g][t][r] + bval;
      }
    }
}

// ---------------------------------------------------------------------------
extern "C" void kernel_launch(void* const* d_in, const int* in_sizes, int n_in,
                              void* d_out, int out_size, void* d_ws, size_t ws_size,
                              hipStream_t stream) {
  const float* q  = (const float*)d_in[0];
  const float* k  = (const float*)d_in[1];
  const float* v  = (const float*)d_in[2];
  // d_in[3] = mask (causal tril, hardcoded), d_in[4] = max_seq_len (== S)
  const float* Wq = (const float*)d_in[5];
  const float* bq = (const float*)d_in[6];
  const float* Wk = (const float*)d_in[7];
  const float* bk = (const float*)d_in[8];
  const float* Wv = (const float*)d_in[9];
  const float* bv = (const float*)d_in[10];
  const float* Wo = (const float*)d_in[11];
  const float* bo = (const float*)d_in[12];

  const size_t seg  = (size_t)M_ * D_;   // 4,194,304 halfs
  const size_t wseg = (size_t)D_ * D_;   // 1,048,576 halfs

  _Float16* base = (_Float16*)d_ws;
  _Float16* qf  = base;                  // f16 copy of q   (later reused as cx)
  _Float16* kf  = qf  + seg;
  _Float16* vf  = kf  + seg;
  _Float16* Wqh = vf  + seg;
  _Float16* Wkh = Wqh + wseg;
  _Float16* Wvh = Wkh + wseg;
  _Float16* Woh = Wvh + wseg;
  _Float16* qh  = Woh + wseg;
  _Float16* kh  = qh  + seg;
  _Float16* vt  = kh  + seg;
  _Float16* cx  = qf;                    // alias: qf dead after kernel 1

  const int nseg  = (int)seg,  gseg  = (int)(seg  / 8 / 256);
  const int nwseg = (int)wseg, gwseg = (int)(wseg / 8 / 256);
  cvt_f32_f16_kernel<<<gseg,  256, 0, stream>>>(q,  qf,  nseg);
  cvt_f32_f16_kernel<<<gseg,  256, 0, stream>>>(k,  kf,  nseg);
  cvt_f32_f16_kernel<<<gseg,  256, 0, stream>>>(v,  vf,  nseg);
  cvt_f32_f16_kernel<<<gwseg, 256, 0, stream>>>(Wq, Wqh, nwseg);
  cvt_f32_f16_kernel<<<gwseg, 256, 0, stream>>>(Wk, Wkh, nwseg);
  cvt_f32_f16_kernel<<<gwseg, 256, 0, stream>>>(Wv, Wvh, nwseg);
  cvt_f32_f16_kernel<<<gwseg, 256, 0, stream>>>(Wo, Woh, nwseg);

  qkv_proj_rope_kernel<<<dim3(D_ / 64, M_ / 256, 3), dim3(256), 0, stream>>>(
      qf, kf, vf, Wqh, Wkh, Wvh, bq, bk, bv, qh, kh, vt);

  flash_attn_kernel<<<dim3((B_ * H_ * S_ / 16) / 4), dim3(128), 0, stream>>>(
      qh, kh, vt, cx);

  out_proj_kernel<<<dim3(D_ / 64, M_ / 256), dim3(256), 0, stream>>>(
      cx, Woh, bo, (float*)d_out);
}
